// EncoderBlock_35802847380030
// MI455X (gfx1250) — compile-verified, matrix operand order checked by reading, not run on hardware
//
#include <hip/hip_runtime.h>
#include <cmath>

typedef _Float16 half_t;
typedef __attribute__((ext_vector_type(16))) _Float16 v16h;
typedef __attribute__((ext_vector_type(8)))  _Float16 v8h;
typedef __attribute__((ext_vector_type(8)))  float    v8f;

#define D_MODEL 768
#define N_HEADS 12
#define DK      64
#define D_FF    3072
#define BB      2
#define SS      2048
#define MROWS   (BB*SS)   // 4096

static __device__ inline v16h pack16(v8h lo, v8h hi) {
  v16h r;
#pragma unroll
  for (int i = 0; i < 8; i++) { r[i] = lo[i]; r[i + 8] = hi[i]; }
  return r;
}

static __device__ inline v8f wmma32(v16h a, v16h b, v8f c) {
  // D = A(16x32 f16) * B(32x16 f16) + C(16x16 f32)
  return __builtin_amdgcn_wmma_f32_16x16x32_f16(false, a, false, b, (short)0, c, false, false);
}

// ---------------- weight convert + transpose: W[KxN] f32 -> WT[NxK] f16 ----------------
__global__ void cvt_w(const float* __restrict__ W, half_t* __restrict__ WT, int K, int N) {
  int idx = blockIdx.x * 256 + threadIdx.x;
  if (idx >= K * N) return;
  int k = idx / N, n = idx % N;
  WT[(size_t)n * K + k] = (half_t)W[(size_t)k * N + n];
}

// ---------------- layernorm row -> f16 ----------------
__global__ __launch_bounds__(256) void ln_f16(const float* __restrict__ x,
                                              const float* __restrict__ g,
                                              const float* __restrict__ be,
                                              half_t* __restrict__ out) {
  __shared__ float rs_[256], rq_[256];
  int row = blockIdx.x;
  const float* xr = x + (size_t)row * D_MODEL;
  float s = 0.f, q = 0.f;
  for (int i = threadIdx.x; i < D_MODEL; i += 256) { float v = xr[i]; s += v; q += v * v; }
  rs_[threadIdx.x] = s; rq_[threadIdx.x] = q; __syncthreads();
  for (int off = 128; off > 0; off >>= 1) {
    if (threadIdx.x < off) {
      rs_[threadIdx.x] += rs_[threadIdx.x + off];
      rq_[threadIdx.x] += rq_[threadIdx.x + off];
    }
    __syncthreads();
  }
  float mu  = rs_[0] * (1.0f / D_MODEL);
  float var = rq_[0] * (1.0f / D_MODEL) - mu * mu;
  float inv = rsqrtf(var + 1e-6f);
  for (int i = threadIdx.x; i < D_MODEL; i += 256)
    out[(size_t)row * D_MODEL + i] = (half_t)((xr[i] - mu) * inv * g[i] + be[i]);
}

// ---------------- WMMA GEMM: out = act(A[MxK]f16 * WT[NxK]^T + bias) (+resid) ----------------
// block = 256 thr = 8 waves; block tile 128x128; K-step 32; double-buffered LDS.
// wave (w&3) -> 32-row group, (w>>2) -> 64-col group: 8 WMMAs per K-step per wave.
__global__ __launch_bounds__(256) void gemm_wmma(
    const half_t* __restrict__ A, const half_t* __restrict__ WT,
    const float* __restrict__ bias, const float* __restrict__ resid,
    float* __restrict__ outF, half_t* __restrict__ outH,
    int M, int N, int K, int relu) {
  __shared__ half_t As[2][128][32] __attribute__((aligned(64)));  // 16 KB
  __shared__ half_t Bs[2][128][32] __attribute__((aligned(64)));  // 16 KB
  int tid = threadIdx.x;
  int lane = tid & 31, wave = tid >> 5;
  int m0 = blockIdx.x * 128, n0 = blockIdx.y * 128;
  int wm = (wave & 3) * 32;
  int wn = (wave >> 2) * 64;
  int r15 = lane & 15;
  int kb  = (lane >> 4) * 8;
  int ar  = tid >> 1, ah = (tid & 1) * 16;   // staging coords: 32B per thread per tile
  v8f acc[2][4] = {};

  // preload K-step 0 into buffer 0
  *(v16h*)&As[0][ar][ah] = *(const v16h*)&A[(size_t)(m0 + ar) * K + ah];
  *(v16h*)&Bs[0][ar][ah] = *(const v16h*)&WT[(size_t)(n0 + ar) * K + ah];

  int nsteps = K / 32;
  for (int s = 0; s < nsteps; ++s) {
    __syncthreads();  // buf(cur) stores visible; all waves done reading buf(nxt)
    if (s + 1 < nsteps) {
      int k0 = (s + 1) * 32, nb = (s + 1) & 1;
      *(v16h*)&As[nb][ar][ah] = *(const v16h*)&A[(size_t)(m0 + ar) * K + k0 + ah];
      *(v16h*)&Bs[nb][ar][ah] = *(const v16h*)&WT[(size_t)(n0 + ar) * K + k0 + ah];
    }
    int cur = s & 1;
    v16h a0 = pack16(*(const v8h*)&As[cur][wm + r15][kb],
                     *(const v8h*)&As[cur][wm + r15][16 + kb]);
    v16h a1 = pack16(*(const v8h*)&As[cur][wm + 16 + r15][kb],
                     *(const v8h*)&As[cur][wm + 16 + r15][16 + kb]);
#pragma unroll
    for (int j = 0; j < 4; j++) {
      int n = wn + j * 16 + r15;
      v16h b = pack16(*(const v8h*)&Bs[cur][n][kb], *(const v8h*)&Bs[cur][n][16 + kb]);
      acc[0][j] = wmma32(a0, b, acc[0][j]);
      acc[1][j] = wmma32(a1, b, acc[1][j]);
    }
  }

  int col = r15;
  int rb  = (lane >> 4) * 8;
#pragma unroll
  for (int rg = 0; rg < 2; rg++) {
#pragma unroll
    for (int j = 0; j < 4; j++) {
      int gn = n0 + wn + j * 16 + col;
      float bv = bias[gn];
#pragma unroll
      for (int r = 0; r < 8; r++) {
        int gm = m0 + wm + rg * 16 + rb + r;
        float v = acc[rg][j][r] + bv;
        if (relu) v = v > 0.f ? v : 0.f;
        if (resid) v += resid[(size_t)gm * N + gn];
        if (outF) outF[(size_t)gm * N + gn] = v;
        if (outH) outH[(size_t)gm * N + gn] = (half_t)v;
      }
    }
  }
}

// ---------------- V transpose: [B,S,D] f16 -> [B,H,64,S] f16 ----------------
__global__ void vtrans(const half_t* __restrict__ v, half_t* __restrict__ vT) {
  int idx = blockIdx.x * 256 + threadIdx.x;
  if (idx >= BB * SS * D_MODEL) return;
  int c = idx % D_MODEL;
  int s = (idx / D_MODEL) % SS;
  int b = idx / (D_MODEL * SS);
  int h = c / DK, d = c % DK;
  vT[((size_t)(b * N_HEADS + h) * DK + d) * SS + s] = v[idx];
}

// ---------------- attention: 8 waves cooperate on one 16-row q tile ----------------
// LDS: sc   = 16 x 2048 f32 scores (128 KB)  [reused as split-K reduction buffer]
//      pf   = 16 x 2048 f16 probabilities (64 KB)
// pass1: waves split the 128 k-tiles (16 each). pass2: 256 threads, exact softmax.
// pass3: waves split the 64 K-steps of P*V (8 each), partials reduced via LDS.
__global__ __launch_bounds__(256) void attn_kernel(
    const half_t* __restrict__ q, const half_t* __restrict__ k,
    const half_t* __restrict__ vT, const int* __restrict__ mask,
    half_t* __restrict__ ctx) {
  extern __shared__ char smem_raw[];
  float*  sc = (float*)smem_raw;            // 16*SS floats
  half_t* pf = (half_t*)(sc + 16 * SS);     // 16*SS halves
  float*  red = sc;                         // pass-3 partial buffer (reuses score region)
  __shared__ float rmax[16][16];
  __shared__ float rsum[16][16];

  int tid  = threadIdx.x;
  int lane = tid & 31, wave = tid >> 5;
  int qt = blockIdx.x;   // 0..127
  int h  = blockIdx.y;
  int b  = blockIdx.z;
  int q0 = qt * 16;
  int r15 = lane & 15;
  int kb  = (lane >> 4) * 8;
  int rb  = (lane >> 4) * 8;

  const half_t* Qr = q + ((size_t)(b * SS + q0 + r15)) * D_MODEL + h * DK;
  v16h aq0 = pack16(*(const v8h*)&Qr[kb],      *(const v8h*)&Qr[16 + kb]);
  v16h aq1 = pack16(*(const v8h*)&Qr[32 + kb], *(const v8h*)&Qr[48 + kb]);

  // ---- pass 1: scores = Q K^T / 8, masked; wave w covers k-tiles [16w, 16w+16) ----
  for (int kk = 0; kk < 16; ++kk) {
    int kt = wave * 16 + kk;
    int kj = kt * 16 + r15;
    const half_t* Kr = k + ((size_t)(b * SS + kj)) * D_MODEL + h * DK;
    v16h bk0 = pack16(*(const v8h*)&Kr[kb],      *(const v8h*)&Kr[16 + kb]);
    v16h bk1 = pack16(*(const v8h*)&Kr[32 + kb], *(const v8h*)&Kr[48 + kb]);
    v8f c = {};
    c = wmma32(aq0, bk0, c);
    c = wmma32(aq1, bk1, c);
    int mk = mask[b * SS + kj];
#pragma unroll
    for (int r = 0; r < 8; r++)
      sc[(size_t)(rb + r) * SS + kt * 16 + r15] = mk ? c[r] * 0.125f : -INFINITY;
  }
  __syncthreads();

  // ---- pass 2: exact softmax; thread = (row = tid>>4, seg = tid&15), 128 cols each ----
  {
    int row = tid >> 4, seg = tid & 15;
    const float* pr = sc + (size_t)row * SS + seg * 128;
    float m = -INFINITY;
    for (int i = 0; i < 128; i++) m = fmaxf(m, pr[i]);
    rmax[row][seg] = m;
    __syncthreads();
    float rm = -INFINITY;
#pragma unroll
    for (int t = 0; t < 16; t++) rm = fmaxf(rm, rmax[row][t]);
    float s = 0.f;
    for (int i = 0; i < 128; i++) s += __expf(pr[i] - rm);
    rsum[row][seg] = s;
    __syncthreads();
    float tot = 0.f;
#pragma unroll
    for (int t = 0; t < 16; t++) tot += rsum[row][t];
    float inv = 1.0f / tot;
    half_t* pw = pf + (size_t)row * SS + seg * 128;
    for (int i = 0; i < 128; i++) pw[i] = (half_t)(__expf(pr[i] - rm) * inv);
  }
  __syncthreads();

  // ---- pass 3: ctx partial = P[:, 256w..] V[256w.., :]; wave w covers K-steps [8w, 8w+8) ----
  v8f acc[4] = {};
  for (int kk = 0; kk < 8; ++kk) {
    int kt2 = wave * 8 + kk;
    const half_t* Pr = pf + (size_t)r15 * SS + kt2 * 32;
    v16h ap = pack16(*(const v8h*)&Pr[kb], *(const v8h*)&Pr[16 + kb]);
#pragma unroll
    for (int j = 0; j < 4; j++) {
      const half_t* Vr = vT + ((size_t)((b * N_HEADS + h) * DK + j * 16 + r15)) * SS + kt2 * 32;
      v16h bv = pack16(*(const v8h*)&Vr[kb], *(const v8h*)&Vr[16 + kb]);
      acc[j] = wmma32(ap, bv, acc[j]);
    }
  }
  // write 16x64 partial tile (scores region is dead now)
#pragma unroll
  for (int j = 0; j < 4; j++)
#pragma unroll
    for (int r = 0; r < 8; r++)
      red[wave * 1024 + (rb + r) * 64 + j * 16 + r15] = acc[j][r];
  __syncthreads();

  // ---- cross-wave reduction + store ctx (f16) ----
#pragma unroll
  for (int t = 0; t < 4; t++) {
    int idx = t * 256 + tid;          // 0..1023
    int row16 = idx >> 6, c64 = idx & 63;
    float s = 0.f;
#pragma unroll
    for (int w = 0; w < 8; w++) s += red[w * 1024 + idx];
    ctx[((size_t)(b * SS + q0 + row16)) * D_MODEL + h * DK + c64] = (half_t)s;
  }
}

extern "C" void kernel_launch(void* const* d_in, const int* in_sizes, int n_in,
                              void* d_out, int out_size, void* d_ws, size_t ws_size,
                              hipStream_t stream) {
  const float* x   = (const float*)d_in[0];
  const int*   msk = (const int*)d_in[1];
  const float* Wq  = (const float*)d_in[2];
  const float* bq  = (const float*)d_in[3];
  const float* Wk  = (const float*)d_in[4];
  const float* bk  = (const float*)d_in[5];
  const float* Wv  = (const float*)d_in[6];
  const float* bv  = (const float*)d_in[7];
  const float* Wo  = (const float*)d_in[8];
  const float* bo  = (const float*)d_in[9];
  const float* g1  = (const float*)d_in[10];
  const float* be1 = (const float*)d_in[11];
  const float* W1  = (const float*)d_in[12];
  const float* b1  = (const float*)d_in[13];
  const float* W2  = (const float*)d_in[14];
  const float* b2  = (const float*)d_in[15];
  const float* g2  = (const float*)d_in[16];
  const float* be2 = (const float*)d_in[17];
  float* out = (float*)d_out;

  char* ws = (char*)d_ws;
  size_t off = 0;
  auto alloc = [&](size_t bytes) {
    char* p = ws + off;
    off += (bytes + 255) & ~(size_t)255;
    return p;
  };
  const size_t actH = (size_t)MROWS * D_MODEL * sizeof(half_t);
  half_t* hf16   = (half_t*)alloc(actH);
  half_t* qf16   = (half_t*)alloc(actH);
  half_t* kf16   = (half_t*)alloc(actH);
  half_t* vf16   = (half_t*)alloc(actH);
  half_t* vTf16  = (half_t*)alloc(actH);
  half_t* ctx16  = (half_t*)alloc(actH);
  float*  x1     = (float*)alloc((size_t)MROWS * D_MODEL * sizeof(float));
  half_t* h2f16  = (half_t*)alloc(actH);
  half_t* ff1    = (half_t*)alloc((size_t)MROWS * D_FF * sizeof(half_t));
  half_t* WqT    = (half_t*)alloc((size_t)D_MODEL * D_MODEL * 2);
  half_t* WkT    = (half_t*)alloc((size_t)D_MODEL * D_MODEL * 2);
  half_t* WvT    = (half_t*)alloc((size_t)D_MODEL * D_MODEL * 2);
  half_t* WoT    = (half_t*)alloc((size_t)D_MODEL * D_MODEL * 2);
  half_t* W1T    = (half_t*)alloc((size_t)D_MODEL * D_FF * 2);
  half_t* W2T    = (half_t*)alloc((size_t)D_FF * D_MODEL * 2);

  // 1) weights -> f16 transposed [NxK]
  {
    int n_dd = D_MODEL * D_MODEL, n_df = D_MODEL * D_FF;
    cvt_w<<<(n_dd + 255) / 256, 256, 0, stream>>>(Wq, WqT, D_MODEL, D_MODEL);
    cvt_w<<<(n_dd + 255) / 256, 256, 0, stream>>>(Wk, WkT, D_MODEL, D_MODEL);
    cvt_w<<<(n_dd + 255) / 256, 256, 0, stream>>>(Wv, WvT, D_MODEL, D_MODEL);
    cvt_w<<<(n_dd + 255) / 256, 256, 0, stream>>>(Wo, WoT, D_MODEL, D_MODEL);
    cvt_w<<<(n_df + 255) / 256, 256, 0, stream>>>(W1, W1T, D_MODEL, D_FF);
    cvt_w<<<(n_df + 255) / 256, 256, 0, stream>>>(W2, W2T, D_FF, D_MODEL);
  }

  // 2) h = LN1(x) (f16)
  ln_f16<<<MROWS, 256, 0, stream>>>(x, g1, be1, hf16);

  // 3) q/k/v projections
  {
    dim3 g(MROWS / 128, D_MODEL / 128);
    gemm_wmma<<<g, 256, 0, stream>>>(hf16, WqT, bq, nullptr, nullptr, qf16,
                                     MROWS, D_MODEL, D_MODEL, 0);
    gemm_wmma<<<g, 256, 0, stream>>>(hf16, WkT, bk, nullptr, nullptr, kf16,
                                     MROWS, D_MODEL, D_MODEL, 0);
    gemm_wmma<<<g, 256, 0, stream>>>(hf16, WvT, bv, nullptr, nullptr, vf16,
                                     MROWS, D_MODEL, D_MODEL, 0);
  }

  // 4) V transpose for contiguous PV B-operand loads
  {
    int tot = BB * SS * D_MODEL;
    vtrans<<<(tot + 255) / 256, 256, 0, stream>>>(vf16, vTf16);
  }

  // 5) attention (192 KB dynamic LDS per 8-wave block; WGP has 320 KB)
  {
    int lds = 16 * SS * (int)sizeof(float) + 16 * SS * (int)sizeof(half_t);
    (void)hipFuncSetAttribute(reinterpret_cast<const void*>(attn_kernel),
                              hipFuncAttributeMaxDynamicSharedMemorySize, lds);
    dim3 ga(SS / 16, N_HEADS, BB);
    attn_kernel<<<ga, 256, lds, stream>>>(qf16, kf16, vTf16, msk, ctx16);
  }

  // 6) x1 = x + ctx @ Wo + bo   (f32, consumed by LN2 and final residual)
  {
    dim3 g(MROWS / 128, D_MODEL / 128);
    gemm_wmma<<<g, 256, 0, stream>>>(ctx16, WoT, bo, x, x1, nullptr,
                                     MROWS, D_MODEL, D_MODEL, 0);
  }

  // 7) h2 = LN2(x1) (f16)
  ln_f16<<<MROWS, 256, 0, stream>>>(x1, g2, be2, h2f16);

  // 8) ff1 = relu(h2 @ W1 + b1) (f16)
  {
    dim3 g(MROWS / 128, D_FF / 128);
    gemm_wmma<<<g, 256, 0, stream>>>(h2f16, W1T, b1, nullptr, nullptr, ff1,
                                     MROWS, D_FF, D_MODEL, 1);
  }

  // 9) out = x1 + ff1 @ W2 + b2 (f32)
  {
    dim3 g(MROWS / 128, D_MODEL / 128);
    gemm_wmma<<<g, 256, 0, stream>>>(ff1, W2T, b2, x1, out, nullptr,
                                     MROWS, D_MODEL, D_FF, 0);
  }
  (void)in_sizes; (void)n_in; (void)out_size; (void)ws_size;
}
